// LocalSelfAttention2d_31782757990567
// MI455X (gfx1250) — matrix operand order repaired and checked by reading, not compiled, not run on hardware
//
#include <hip/hip_runtime.h>
#include <hip/hip_bf16.h>

// Problem constants (match reference)
#define BATCH 16
#define CH    256
#define HH    64
#define WW    64
#define HWSZ  4096
#define HEADS 8
#define DH    32
#define NWIN  64   // windows per (b, head): 8x8
#define PP    64   // positions per window: 8x8

typedef __attribute__((ext_vector_type(16))) __bf16 bf16x16;
typedef __attribute__((ext_vector_type(8)))  float  f32x8;
typedef __attribute__((ext_vector_type(4)))  unsigned int u32x4;

union BfOp { u32x4 q[2]; bf16x16 v; };

// f32 pair -> packed bf16, round-to-nearest: one add per value + one v_perm_b32.
__device__ __forceinline__ unsigned pk2(float a, float b) {
  unsigned ua = __float_as_uint(a) + 0x8000u;
  unsigned ub = __float_as_uint(b) + 0x8000u;
  return __builtin_amdgcn_perm(ub, ua, 0x07060302u);  // {ub.hi16, ua.hi16}
}
__device__ __forceinline__ unsigned short f2b(float a) {
  return (unsigned short)((__float_as_uint(a) + 0x8000u) >> 16);
}

// ---------------------------------------------------------------------------
// Kernel 0: convert both weight matrices to bf16 once (row-major [o][c]).
// ---------------------------------------------------------------------------
__global__ __launch_bounds__(256) void cvt_w_kernel(
    const float* __restrict__ wp, const float* __restrict__ wo,
    unsigned short* __restrict__ wp16, unsigned short* __restrict__ wo16) {
  const int i = blockIdx.x * 256 + threadIdx.x;  // pair index
  if (i < 98304) {                               // 768*256/2
    float2 f = *(const float2*)(wp + 2 * i);
    ((unsigned*)wp16)[i] = pk2(f.x, f.y);
  } else {                                       // 256*256/2 pairs
    const int j = i - 98304;
    float2 f = *(const float2*)(wo + 2 * j);
    ((unsigned*)wo16)[j] = pk2(f.x, f.y);
  }
}

// ---------------------------------------------------------------------------
// Kernel 0b: x f32 [b][c][h][w] -> xt bf16 [b][posW][c] (window order,
// channel-contiguous). One block per 64pos x 64c tile, LDS transpose.
// After this, x is never touched again and GEMM B-operands are b128 loads.
// ---------------------------------------------------------------------------
__global__ __launch_bounds__(256) void xt_kernel(
    const float* __restrict__ x, unsigned short* __restrict__ xt) {
  __shared__ unsigned short t[64 * 66];  // [pp][c], padded row to kill conflicts
  const int tid = threadIdx.x;
  int idx = blockIdx.x;
  const int ct = idx & 3;  idx >>= 2;    // 256/64 channel tiles
  const int wi = idx & 63; idx >>= 6;    // 64 windows (posW tile == window)
  const int b  = idx;
  const int c0 = ct * 64;

  // read coalesced along pos, convert, LDS store packed pairs of c
  const int pp = tid & 63;
  const int xh = (wi >> 3) * 8 + (pp >> 3);
  const int xw = (wi & 7) * 8 + (pp & 7);
  const float* xb = x + (size_t)b * CH * HWSZ + xh * WW + xw;
  #pragma unroll
  for (int i = 0; i < 8; ++i) {
    const int cl = i * 8 + (tid >> 6) * 2;
    const float lo = xb[(size_t)(c0 + cl) * HWSZ];
    const float hi = xb[(size_t)(c0 + cl + 1) * HWSZ];
    *(unsigned*)&t[pp * 66 + cl] = pk2(lo, hi);
  }
  __syncthreads();
  // write coalesced along c
  const int c2 = (tid & 31) * 2;
  #pragma unroll
  for (int i = 0; i < 8; ++i) {
    const int p2 = i * 8 + (tid >> 5);
    *(unsigned*)&xt[((size_t)b * HWSZ + wi * PP + p2) * CH + c0 + c2] =
        *(const unsigned*)&t[p2 * 66 + c2];
  }
}

// ---------------------------------------------------------------------------
// Kernel 1: QKV projection GEMM, LDS-free, K-loop double-buffered.
// out[o,pos] = sum_c wp[o,c]*x[c,pos]. Wave: 64o x 16pos (4 C tiles);
// block = 8 waves = 64o x 128pos. A and B loaded straight from global bf16
// in WMMA lane layout (2x b128 each); next K-step prefetched over WMMAs.
// ---------------------------------------------------------------------------
__global__ __launch_bounds__(256) void qkv_gemm_kernel(
    const unsigned short* __restrict__ xt, const unsigned short* __restrict__ wp16,
    unsigned short* __restrict__ qws, unsigned short* __restrict__ kws,
    unsigned short* __restrict__ vws) {
  const int tid  = threadIdx.x;
  const int lane = tid & 31, wv = tid >> 5;
  const int half = lane >> 4, col = lane & 15;

  int idx = blockIdx.x;
  const int ob = idx % 12; idx /= 12;      // 768 / 64
  const int pb = idx % 32; idx /= 32;      // 4096 / 128
  const int b  = idx;
  const int o0 = ob * 64;

  const int pos = pb * 128 + wv * 16 + col;
  const int wi = pos >> 6, pp = pos & 63;
  const unsigned short* xrow = xt + ((size_t)b * HWSZ + pos) * CH;
  const unsigned short* wbase = wp16 + (size_t)(o0 + col) * CH;

  BfOp bm[2], am[2][4];
  auto load_step = [&](int kk, int buf) {
    const int c0 = kk * 32;
    const unsigned short* bp = xrow + c0 + half * 8;
    bm[buf].q[0] = *(const u32x4*)bp;
    bm[buf].q[1] = *(const u32x4*)(bp + 16);
    #pragma unroll
    for (int m = 0; m < 4; ++m) {
      const unsigned short* ap = wbase + (size_t)(m * 16) * CH + c0 + half * 8;
      am[buf][m].q[0] = *(const u32x4*)ap;
      am[buf][m].q[1] = *(const u32x4*)(ap + 16);
    }
  };

  f32x8 acc[4] = {};
  load_step(0, 0);
  #pragma unroll
  for (int kk = 0; kk < 8; ++kk) {
    const int cur = kk & 1;
    if (kk < 7) load_step(kk + 1, cur ^ 1);   // prefetch next K-step
    #pragma unroll
    for (int m = 0; m < 4; ++m)
      acc[m] = __builtin_amdgcn_wmma_f32_16x16x32_bf16(
          false, am[cur][m].v, false, bm[cur].v, (short)0, acc[m], false, false);
  }

  // Epilogue. t is uniform across the block.
  const int t = o0 >> 8;  // 0=Q 1=K 2=V
  if (t < 2) {
    unsigned short* __restrict__ dst = (t == 0) ? qws : kws;
    #pragma unroll
    for (int m = 0; m < 4; ++m) {
      const int o = o0 + m * 16 + 8 * half;  // 8 consecutive channels
      const int hd = (o >> 5) & 7, d0 = o & 31;
      const int wbh = (b * HEADS + hd) * NWIN + wi;
      const u32x4 pv = {pk2(acc[m][0], acc[m][1]), pk2(acc[m][2], acc[m][3]),
                        pk2(acc[m][4], acc[m][5]), pk2(acc[m][6], acc[m][7])};
      *(u32x4*)&dst[((size_t)wbh * PP + pp) * DH + d0] = pv;
    }
  } else {
    #pragma unroll
    for (int m = 0; m < 4; ++m) {
      #pragma unroll
      for (int r = 0; r < 8; ++r) {  // lanes cover consecutive pp: coalesced
        const int o = o0 + m * 16 + r + 8 * half;
        const int hd = (o >> 5) & 7, d = o & 31;
        const int wbh = (b * HEADS + hd) * NWIN + wi;
        vws[((size_t)wbh * DH + d) * PP + pp] = f2b(acc[m][r]);
      }
    }
  }
}

// ---------------------------------------------------------------------------
// Kernel 2: windowed attention, one block per (b, head, window).
// Scores = 16 WMMA tiles (single K=32 WMMA, D==32), scale + relative bias,
// softmax via LDS, attn@V = 8 tiles (2 chained WMMAs). Output packed bf16
// [b][posW][ci] via LDS transpose (reusing the scores buffer).
// ---------------------------------------------------------------------------
__global__ __launch_bounds__(256) void win_attn_kernel(
    const unsigned short* __restrict__ qws, const unsigned short* __restrict__ kws,
    const unsigned short* __restrict__ vws, const float* __restrict__ position,
    unsigned short* __restrict__ oact) {
  __shared__ float sS[PP * PP];           // 16 KB scores / exp / O-staging
  __shared__ unsigned short sP[PP * PP];  // 8 KB probs (bf16)
  const int tid  = threadIdx.x;
  const int lane = tid & 31, wv = tid >> 5;
  const int half = lane >> 4, col = lane & 15;

  int idx = blockIdx.x;
  const int wi = idx % NWIN; idx /= NWIN;
  const int hd = idx % HEADS; idx /= HEADS;
  const int b  = idx;

  const size_t base = (size_t)((b * HEADS + hd) * NWIN + wi);
  const unsigned short* qb = qws + base * PP * DH;   // [pp][d]
  const unsigned short* kb = kws + base * PP * DH;   // [pp][d]
  const unsigned short* vb = vws + base * DH * PP;   // [d][pp]
  const float* posb = position + hd * 256;           // [16][16]
  const float scale = 0.17677669529663688f;          // 1/sqrt(32)

  // ---- scores: tiles (qi,ki), 2 per wave; operands preloaded, then WMMAs ----
  {
    BfOp am[2], bm[2];
    #pragma unroll
    for (int t = 0; t < 2; ++t) {
      const int ti = wv * 2 + t, qi = ti >> 2, ki = ti & 3;
      const unsigned short* ap = qb + (qi * 16 + col) * DH + half * 8;
      am[t].q[0] = *(const u32x4*)ap;
      am[t].q[1] = *(const u32x4*)(ap + 16);
      const unsigned short* bp = kb + (ki * 16 + col) * DH + half * 8;
      bm[t].q[0] = *(const u32x4*)bp;
      bm[t].q[1] = *(const u32x4*)(bp + 16);
    }
    f32x8 c[2];
    #pragma unroll
    for (int t = 0; t < 2; ++t) {
      f32x8 z = {};
      c[t] = __builtin_amdgcn_wmma_f32_16x16x32_bf16(false, am[t].v, false,
                                                     bm[t].v, (short)0, z,
                                                     false, false);
    }
    #pragma unroll
    for (int t = 0; t < 2; ++t) {
      const int ti = wv * 2 + t, qi = ti >> 2, ki = ti & 3;
      const int k = ki * 16 + col, kr = k >> 3, kc = k & 7;
      #pragma unroll
      for (int r = 0; r < 8; ++r) {
        const int q = qi * 16 + r + 8 * half;
        const int qr = q >> 3, qc = q & 7;
        sS[q * PP + k] =
            c[t][r] * scale + posb[(kr - qr + 8) * 16 + (kc - qc + 8)];
      }
    }
  }
  __syncthreads();

  // ---- softmax over k: one row per thread (threads 0..63) ----
  if (tid < PP) {
    float mx = -3.0e38f;
    for (int j = 0; j < PP; ++j) mx = fmaxf(mx, sS[tid * PP + j]);
    float s = 0.f;
    for (int j = 0; j < PP; ++j) {
      float e = __expf(sS[tid * PP + j] - mx);
      sS[tid * PP + j] = e;
      s += e;
    }
    const float inv = 1.f / s;
    for (int j = 0; j < PP; j += 2)
      *(unsigned*)&sP[tid * PP + j] =
          pk2(sS[tid * PP + j] * inv, sS[tid * PP + j + 1] * inv);
  }
  __syncthreads();

  // ---- O = P @ V : 8 tiles (qi x di), one per wave, K = 64 = 2 WMMAs ----
  const int qi = wv >> 1, di = wv & 1;
  f32x8 acc = {};
  {
    BfOp am[2], bm[2];
    #pragma unroll
    for (int kc2 = 0; kc2 < 2; ++kc2) {  // preload both K-chunks
      const unsigned short* ap = &sP[(qi * 16 + col) * PP + kc2 * 32 + half * 8];
      am[kc2].q[0] = *(const u32x4*)ap;
      am[kc2].q[1] = *(const u32x4*)(ap + 16);
      const unsigned short* bp =
          vb + (di * 16 + col) * PP + kc2 * 32 + half * 8;
      bm[kc2].q[0] = *(const u32x4*)bp;
      bm[kc2].q[1] = *(const u32x4*)(bp + 16);
    }
    #pragma unroll
    for (int kc2 = 0; kc2 < 2; ++kc2)
      acc = __builtin_amdgcn_wmma_f32_16x16x32_bf16(
          false, am[kc2].v, false, bm[kc2].v, (short)0, acc, false, false);
  }
  // stage O tile [64q][32d] f32 in LDS (reuse sS), then pack -> one b128/thread
  {
    float* sO = sS;
    const int dl = di * 16 + col;
    #pragma unroll
    for (int r = 0; r < 8; ++r) sO[(qi * 16 + r + 8 * half) * DH + dl] = acc[r];
    __syncthreads();
    const int q = tid >> 2, dc = (tid & 3) * 8;
    const float* s = &sO[q * DH + dc];
    const u32x4 pv = {pk2(s[0], s[1]), pk2(s[2], s[3]),
                      pk2(s[4], s[5]), pk2(s[6], s[7])};
    *(u32x4*)&oact[((size_t)b * HWSZ + wi * PP + q) * CH + hd * DH + dc] = pv;
  }
}

// ---------------------------------------------------------------------------
// Kernel 3: output projection GEMM + bias, LDS-free, double-buffered;
// un-windows posW -> (h,w) on store.
// ---------------------------------------------------------------------------
__global__ __launch_bounds__(256) void out_gemm_kernel(
    const unsigned short* __restrict__ oact, const unsigned short* __restrict__ wo16,
    const float* __restrict__ bout, float* __restrict__ out) {
  const int tid  = threadIdx.x;
  const int lane = tid & 31, wv = tid >> 5;
  const int half = lane >> 4, col = lane & 15;

  int idx = blockIdx.x;
  const int ob = idx % 4;  idx /= 4;       // 256 / 64
  const int pb = idx % 32; idx /= 32;      // 4096 / 128
  const int b  = idx;
  const int o0 = ob * 64;

  const int pos = pb * 128 + wv * 16 + col;
  const int wi = pos >> 6, pp = pos & 63;
  const int xh = (wi >> 3) * 8 + (pp >> 3);
  const int xw = (wi & 7) * 8 + (pp & 7);
  const unsigned short* xrow = oact + ((size_t)b * HWSZ + pos) * CH;
  const unsigned short* wbase = wo16 + (size_t)(o0 + col) * CH;

  BfOp bm[2], am[2][4];
  auto load_step = [&](int kk, int buf) {
    const int c0 = kk * 32;
    const unsigned short* bp = xrow + c0 + half * 8;
    bm[buf].q[0] = *(const u32x4*)bp;
    bm[buf].q[1] = *(const u32x4*)(bp + 16);
    #pragma unroll
    for (int m = 0; m < 4; ++m) {
      const unsigned short* ap = wbase + (size_t)(m * 16) * CH + c0 + half * 8;
      am[buf][m].q[0] = *(const u32x4*)ap;
      am[buf][m].q[1] = *(const u32x4*)(ap + 16);
    }
  };

  f32x8 acc[4] = {};
  load_step(0, 0);
  #pragma unroll
  for (int kk = 0; kk < 8; ++kk) {
    const int cur = kk & 1;
    if (kk < 7) load_step(kk + 1, cur ^ 1);
    #pragma unroll
    for (int m = 0; m < 4; ++m)
      acc[m] = __builtin_amdgcn_wmma_f32_16x16x32_bf16(
          false, am[cur][m].v, false, bm[cur].v, (short)0, acc[m], false, false);
  }

  #pragma unroll
  for (int m = 0; m < 4; ++m) {
    const int o = o0 + m * 16 + 8 * half;   // 8 consecutive channels
    const float4 bz0 = *(const float4*)&bout[o];
    const float4 bz1 = *(const float4*)&bout[o + 4];
    float* op = out + (size_t)b * CH * HWSZ + (size_t)o * HWSZ + xh * WW + xw;
    op[0 * HWSZ] = acc[m][0] + bz0.x;
    op[1 * HWSZ] = acc[m][1] + bz0.y;
    op[2 * HWSZ] = acc[m][2] + bz0.z;
    op[3 * HWSZ] = acc[m][3] + bz0.w;
    op[4 * HWSZ] = acc[m][4] + bz1.x;
    op[5 * HWSZ] = acc[m][5] + bz1.y;
    op[6 * HWSZ] = acc[m][6] + bz1.z;
    op[7 * HWSZ] = acc[m][7] + bz1.w;
  }
}

// ---------------------------------------------------------------------------
extern "C" void kernel_launch(void* const* d_in, const int* in_sizes, int n_in,
                              void* d_out, int out_size, void* d_ws, size_t ws_size,
                              hipStream_t stream) {
  const float* x        = (const float*)d_in[0];  // [16,256,64,64]
  const float* w_proj   = (const float*)d_in[1];  // [768,256]
  const float* position = (const float*)d_in[2];  // [8,16,16]
  const float* w_out    = (const float*)d_in[3];  // [256,256]
  const float* b_out    = (const float*)d_in[4];  // [256]
  float* out = (float*)d_out;

  // workspace layout (bytes):
  //   wp16 bf16 [768][256]            :   384 KiB
  //   wo16 bf16 [256][256]            :   128 KiB
  //   xt   bf16 [b][posW][c]          :    32 MiB
  //   qws  bf16 [b][head][wi][pp][d]  :    32 MiB
  //   kws  bf16 [b][head][wi][pp][d]  :    32 MiB
  //   vws  bf16 [b][head][wi][d][pp]  :    32 MiB
  //   oact bf16 [b][posW][ci]         :    32 MiB
  char* ws = (char*)d_ws;
  const size_t ACT_BYTES = (size_t)BATCH * HWSZ * CH * 2;  // 33554432
  unsigned short* wp16 = (unsigned short*)(ws);
  unsigned short* wo16 = (unsigned short*)(ws + 393216);
  unsigned short* xt   = (unsigned short*)(ws + 524288);
  unsigned short* qws  = (unsigned short*)(ws + 524288 + 1 * ACT_BYTES);
  unsigned short* kws  = (unsigned short*)(ws + 524288 + 2 * ACT_BYTES);
  unsigned short* vws  = (unsigned short*)(ws + 524288 + 3 * ACT_BYTES);
  unsigned short* oact = (unsigned short*)(ws + 524288 + 4 * ACT_BYTES);

  cvt_w_kernel<<<512, 256, 0, stream>>>(w_proj, w_out, wp16, wo16);
  xt_kernel<<<BATCH * 64 * 4, 256, 0, stream>>>(x, xt);
  qkv_gemm_kernel<<<BATCH * 32 * 12, 256, 0, stream>>>(xt, wp16, qws, kws, vws);
  win_attn_kernel<<<BATCH * HEADS * NWIN, 256, 0, stream>>>(qws, kws, vws,
                                                            position, oact);
  out_gemm_kernel<<<BATCH * 32 * 4, 256, 0, stream>>>(oact, wo16, b_out, out);
}
